// KVPLMStarEncoder_56538949485134
// MI455X (gfx1250) — matrix operand order repaired and checked by reading, not compiled, not run on hardware
//
#include <hip/hip_runtime.h>
#include <hip/hip_bf16.h>
#include <math.h>

typedef __attribute__((ext_vector_type(16))) __bf16 v16bf_t;
typedef __attribute__((ext_vector_type(8)))  float  v8f_t;
typedef unsigned short ushort_t;

// ---------------------------------------------------------------------------
// helpers
// ---------------------------------------------------------------------------
__device__ __forceinline__ ushort_t f2bf(float f) {
    union { __bf16 h; ushort_t u; } c;
    c.h = (__bf16)f;                      // native v_cvt (pairs fuse to v_cvt_pk_bf16_f32)
    return c.u;
}

__device__ __forceinline__ v8f_t wmma_bf16(v16bf_t a, v16bf_t b, v8f_t c) {
    // D = A(16x32 bf16) * B(32x16 bf16) + C(16x16 f32)
    return __builtin_amdgcn_wmma_f32_16x16x32_bf16(
        false, a, false, b, (short)0, c, false, false);
}

// build A/B fragment from two groups of 8 consecutive f32 (global or LDS)
__device__ __forceinline__ v16bf_t frag8x2(const float* p0, const float* p1) {
    const float4 a0 = *(const float4*)(p0);
    const float4 a1 = *(const float4*)(p0 + 4);
    const float4 b0 = *(const float4*)(p1);
    const float4 b1 = *(const float4*)(p1 + 4);
    v16bf_t v;
    v[0]=(__bf16)a0.x;  v[1]=(__bf16)a0.y;  v[2]=(__bf16)a0.z;  v[3]=(__bf16)a0.w;
    v[4]=(__bf16)a1.x;  v[5]=(__bf16)a1.y;  v[6]=(__bf16)a1.z;  v[7]=(__bf16)a1.w;
    v[8]=(__bf16)b0.x;  v[9]=(__bf16)b0.y;  v[10]=(__bf16)b0.z; v[11]=(__bf16)b0.w;
    v[12]=(__bf16)b1.x; v[13]=(__bf16)b1.y; v[14]=(__bf16)b1.z; v[15]=(__bf16)b1.w;
    return v;
}

// build fragment from 16 consecutive f32
__device__ __forceinline__ v16bf_t frag16(const float* p) { return frag8x2(p, p + 8); }

// build fragment from bf16 bits already resident in LDS (two ds_load_b128)
__device__ __forceinline__ v16bf_t frag_lds(const ushort_t* base, int off0, int off1) {
    union { v16bf_t v; uint4 q[2]; } r;
    r.q[0] = *(const uint4*)(base + off0);
    r.q[1] = *(const uint4*)(base + off1);
    return r.v;
}

// ---------------------------------------------------------------------------
// fp32 -> bf16-bits weight conversion
// ---------------------------------------------------------------------------
__global__ void cvt_bf16_kernel(const float* __restrict__ s, ushort_t* __restrict__ d, int n) {
    int i = blockIdx.x * blockDim.x + threadIdx.x;
    int stride = gridDim.x * blockDim.x;
    for (; i < n; i += stride) d[i] = f2bf(s[i]);
}

// ---------------------------------------------------------------------------
// WMMA GEMM: C[M,N] = A[M,K](f32) * B[K,N](bf16 bits) + bias (+gelu)
// block = 256 threads (8 waves), tile 128(M) x 128(N), K-step 32
// wave grid 4(M) x 2(N): each wave owns 32x64 = 8 WMMA accumulators
// EPI: 1 = +bias, 2 = +bias then exact GELU
// ---------------------------------------------------------------------------
#define ASTR 40
#define BSTR 40
template <int EPI>
__global__ __launch_bounds__(256) void gemm_bf16_kernel(
    const float* __restrict__ A, const ushort_t* __restrict__ Bw,
    const float* __restrict__ bias, float* __restrict__ C,
    int M, int N, int K)
{
    __shared__ __align__(16) ushort_t As[128 * ASTR];
    __shared__ __align__(16) ushort_t Bs[128 * BSTR];

    const int tid  = threadIdx.x;
    const int lane = tid & 31;
    const int wave = tid >> 5;
    const int wm = wave & 3;          // 4 waves along M (32 rows each)
    const int wn = wave >> 2;         // 2 waves along N (64 cols each)
    const int m0 = blockIdx.y * 128;
    const int n0 = blockIdx.x * 128;
    const int l15 = lane & 15, lh = lane >> 4;

    v8f_t z = {0.f,0.f,0.f,0.f,0.f,0.f,0.f,0.f};
    v8f_t acc[2][4];
    #pragma unroll
    for (int mi = 0; mi < 2; ++mi)
        #pragma unroll
        for (int ni = 0; ni < 4; ++ni) acc[mi][ni] = z;

    const int arow = tid >> 1;          // 0..127
    const int acol = (tid & 1) * 16;    // 0 / 16
    const int bk   = tid >> 3;          // 0..31
    const int bn   = (tid & 7) * 16;    // 0..112

    for (int kk = 0; kk < K; kk += 32) {
        // stage A tile (convert f32 -> bf16 bits)
        {
            const float* ap = A + (size_t)(m0 + arow) * K + kk + acol;
            float4 f0 = *(const float4*)(ap);
            float4 f1 = *(const float4*)(ap + 4);
            float4 f2 = *(const float4*)(ap + 8);
            float4 f3 = *(const float4*)(ap + 12);
            ushort_t* as = &As[arow * ASTR + acol];
            as[0]=f2bf(f0.x);  as[1]=f2bf(f0.y);  as[2]=f2bf(f0.z);  as[3]=f2bf(f0.w);
            as[4]=f2bf(f1.x);  as[5]=f2bf(f1.y);  as[6]=f2bf(f1.z);  as[7]=f2bf(f1.w);
            as[8]=f2bf(f2.x);  as[9]=f2bf(f2.y);  as[10]=f2bf(f2.z); as[11]=f2bf(f2.w);
            as[12]=f2bf(f3.x); as[13]=f2bf(f3.y); as[14]=f2bf(f3.z); as[15]=f2bf(f3.w);
        }
        // stage B tile transposed: Bs[n][k]  (32 k-rows x 128 n-cols)
        {
            const ushort_t* bp = Bw + (size_t)(kk + bk) * N + n0 + bn;
            union { uint4 q; ushort_t s[8]; } t0, t1;
            t0.q = *(const uint4*)bp;
            t1.q = *(const uint4*)(bp + 8);
            #pragma unroll
            for (int i = 0; i < 8; ++i) Bs[(bn + i) * BSTR + bk] = t0.s[i];
            #pragma unroll
            for (int i = 0; i < 8; ++i) Bs[(bn + 8 + i) * BSTR + bk] = t1.s[i];
            // prefetch next K-step's B tile toward the WGP
            if (kk + 32 < K)
                __builtin_prefetch(bp + (size_t)32 * N, 0, 1);
        }
        __syncthreads();

        v16bf_t af[2], bfr[4];
        #pragma unroll
        for (int mi = 0; mi < 2; ++mi) {
            int r = (wm * 32 + mi * 16 + l15) * ASTR + lh * 8;
            af[mi] = frag_lds(As, r, r + 16);
        }
        #pragma unroll
        for (int ni = 0; ni < 4; ++ni) {
            int c = (wn * 64 + ni * 16 + l15) * BSTR + lh * 8;
            bfr[ni] = frag_lds(Bs, c, c + 16);
        }
        #pragma unroll
        for (int mi = 0; mi < 2; ++mi)
            #pragma unroll
            for (int ni = 0; ni < 4; ++ni)
                acc[mi][ni] = wmma_bf16(af[mi], bfr[ni], acc[mi][ni]);
        __syncthreads();
    }

    // epilogue (EPI compile-time: no branches in the hot store loop)
    #pragma unroll
    for (int mi = 0; mi < 2; ++mi) {
        #pragma unroll
        for (int ni = 0; ni < 4; ++ni) {
            int col = n0 + wn * 64 + ni * 16 + l15;
            float bv = bias[col];
            #pragma unroll
            for (int r = 0; r < 8; ++r) {
                int row = m0 + wm * 32 + mi * 16 + r + lh * 8;
                float v = acc[mi][ni][r] + bv;
                if (EPI == 2) v = 0.5f * v * (1.0f + erff(v * 0.70710678118654752f));
                C[(size_t)row * N + col] = v;
            }
        }
    }
}

// ---------------------------------------------------------------------------
// Attention: one wave handles one (batch, head, 16-query tile).
// scores (16x512) kept in LDS, softmax in-wave, then P@V with WMMA.
// ---------------------------------------------------------------------------
#define SSTR 516
__global__ __launch_bounds__(32) void attn_kernel(
    const float* __restrict__ Q, const float* __restrict__ Kd,
    const float* __restrict__ Vd, const int* __restrict__ mask,
    float* __restrict__ ctx)
{
    __shared__ __align__(16) float    sc[16 * SSTR];
    __shared__ __align__(16) ushort_t Vt[64 * 40];

    const int bx = blockIdx.x;
    const int qt = bx & 31;            // 32 query tiles (512/16)
    const int h  = (bx >> 5) % 12;
    const int bb = bx / (32 * 12);
    const int lane = threadIdx.x;
    const int l15 = lane & 15, lh = lane >> 4;
    const int row0 = bb * 512 + qt * 16;

    v8f_t z = {0.f,0.f,0.f,0.f,0.f,0.f,0.f,0.f};

    // Q fragments (head dim 64 = two K=32 slabs)
    const float* qbase = Q + (size_t)(row0 + l15) * 768 + h * 64;
    v16bf_t aQ0 = frag8x2(qbase + lh * 8,      qbase + lh * 8 + 16);
    v16bf_t aQ1 = frag8x2(qbase + 32 + lh * 8, qbase + 32 + lh * 8 + 16);

    // scores = Q @ K^T * scale + mask bias  -> LDS
    for (int n0 = 0; n0 < 512; n0 += 16) {
        int n = n0 + l15;
        const float* kbase = Kd + (size_t)(bb * 512 + n) * 768 + h * 64;
        v16bf_t bK0 = frag8x2(kbase + lh * 8,      kbase + lh * 8 + 16);
        v16bf_t bK1 = frag8x2(kbase + 32 + lh * 8, kbase + 32 + lh * 8 + 16);
        v8f_t c = z;
        c = wmma_bf16(aQ0, bK0, c);
        c = wmma_bf16(aQ1, bK1, c);
        float mb = (1.0f - (float)mask[bb * 512 + n]) * -10000.0f;
        #pragma unroll
        for (int r = 0; r < 8; ++r)
            sc[(r + lh * 8) * SSTR + n] = c[r] * 0.125f + mb;
    }
    __syncthreads();

    // softmax over 512 keys: lane pair (l15, l15+16) each scans half a row
    {
        int m = l15, base = lh * 256;
        float mx = -3.0e38f;
        for (int j = 0; j < 256; ++j) mx = fmaxf(mx, sc[m * SSTR + base + j]);
        mx = fmaxf(mx, __shfl_xor(mx, 16));
        float sum = 0.f;
        for (int j = 0; j < 256; ++j) {
            float e = __expf(sc[m * SSTR + base + j] - mx);
            sc[m * SSTR + base + j] = e;
            sum += e;
        }
        sum += __shfl_xor(sum, 16);
        float inv = 1.0f / sum;
        for (int j = 0; j < 256; ++j) sc[m * SSTR + base + j] *= inv;
    }
    __syncthreads();

    // ctx = P @ V  (accumulate 16x64 output, K loop over 512 keys, step 32)
    v8f_t acc[4]; acc[0]=z; acc[1]=z; acc[2]=z; acc[3]=z;
    for (int kk = 0; kk < 512; kk += 32) {
        // stage V tile transposed as bf16: Vt[d][key], lane owns key kk+lane
        const float* vrow = Vd + (size_t)(bb * 512 + kk + lane) * 768 + h * 64;
        #pragma unroll
        for (int d4 = 0; d4 < 64; d4 += 4) {
            float4 f = *(const float4*)(vrow + d4);
            Vt[(d4 + 0) * 40 + lane] = f2bf(f.x);
            Vt[(d4 + 1) * 40 + lane] = f2bf(f.y);
            Vt[(d4 + 2) * 40 + lane] = f2bf(f.z);
            Vt[(d4 + 3) * 40 + lane] = f2bf(f.w);
        }
        __syncthreads();
        v16bf_t aP = frag8x2(&sc[l15 * SSTR + kk + lh * 8],
                             &sc[l15 * SSTR + kk + lh * 8 + 16]);
        #pragma unroll
        for (int t = 0; t < 4; ++t) {
            int d = (t * 16 + l15) * 40 + lh * 8;
            v16bf_t bV = frag_lds(Vt, d, d + 16);
            acc[t] = wmma_bf16(aP, bV, acc[t]);
        }
        __syncthreads();
    }
    #pragma unroll
    for (int t = 0; t < 4; ++t)
        #pragma unroll
        for (int r = 0; r < 8; ++r)
            ctx[(size_t)(row0 + r + lh * 8) * 768 + h * 64 + t * 16 + l15] = acc[t][r];
}

// ---------------------------------------------------------------------------
// Embedding gather + add pos/type + LayerNorm (one block per token)
// ---------------------------------------------------------------------------
__global__ __launch_bounds__(256) void embed_ln_kernel(
    const int* __restrict__ ids, const int* __restrict__ tts,
    const float* __restrict__ we, const float* __restrict__ se,
    const float* __restrict__ pe, const float* __restrict__ te,
    const float* __restrict__ g, const float* __restrict__ b,
    float* __restrict__ out)
{
    const int tok = blockIdx.x, tid = threadIdx.x;
    const int ss = tok & 511;
    __shared__ float buf[768];
    __shared__ float red[256];
    __shared__ float red2[256];
    int id = ids[tok];
    const float* e = (id >= 30700) ? (se + (size_t)(id - 30700) * 768)
                                   : (we + (size_t)id * 768);
    int ty = tts[tok];
    float s = 0.f, s2 = 0.f;
    #pragma unroll
    for (int j = 0; j < 3; ++j) {
        int c = tid + j * 256;
        float v = e[c] + pe[(size_t)ss * 768 + c] + te[(size_t)ty * 768 + c];
        buf[c] = v; s += v; s2 += v * v;
    }
    red[tid] = s; red2[tid] = s2; __syncthreads();
    for (int o = 128; o > 0; o >>= 1) {
        if (tid < o) { red[tid] += red[tid + o]; red2[tid] += red2[tid + o]; }
        __syncthreads();
    }
    float mean = red[0] * (1.0f / 768.0f);
    float var  = red2[0] * (1.0f / 768.0f) - mean * mean;
    float inv  = rsqrtf(var + 1e-12f);
    #pragma unroll
    for (int j = 0; j < 3; ++j) {
        int c = tid + j * 256;
        out[(size_t)tok * 768 + c] = (buf[c] - mean) * inv * g[c] + b[c];
    }
}

// ---------------------------------------------------------------------------
// out = LayerNorm(A + R)  (one block per row)
// ---------------------------------------------------------------------------
__global__ __launch_bounds__(256) void ln_res_kernel(
    const float* __restrict__ A, const float* __restrict__ R,
    const float* __restrict__ g, const float* __restrict__ b,
    float* __restrict__ out)
{
    const int row = blockIdx.x, tid = threadIdx.x;
    __shared__ float buf[768];
    __shared__ float red[256];
    __shared__ float red2[256];
    float s = 0.f, s2 = 0.f;
    #pragma unroll
    for (int j = 0; j < 3; ++j) {
        int c = tid + j * 256;
        float v = A[(size_t)row * 768 + c] + R[(size_t)row * 768 + c];
        buf[c] = v; s += v; s2 += v * v;
    }
    red[tid] = s; red2[tid] = s2; __syncthreads();
    for (int o = 128; o > 0; o >>= 1) {
        if (tid < o) { red[tid] += red[tid + o]; red2[tid] += red2[tid + o]; }
        __syncthreads();
    }
    float mean = red[0] * (1.0f / 768.0f);
    float var  = red2[0] * (1.0f / 768.0f) - mean * mean;
    float inv  = rsqrtf(var + 1e-12f);
    #pragma unroll
    for (int j = 0; j < 3; ++j) {
        int c = tid + j * 256;
        out[(size_t)row * 768 + c] = (buf[c] - mean) * inv * g[c] + b[c];
    }
}

// ---------------------------------------------------------------------------
// host orchestration
// ---------------------------------------------------------------------------
extern "C" void kernel_launch(void* const* d_in, const int* in_sizes, int n_in,
                              void* d_out, int out_size, void* d_ws, size_t ws_size,
                              hipStream_t stream) {
    const int*   input_ids = (const int*)  d_in[0];
    const int*   attn_mask = (const int*)  d_in[1];
    const int*   type_ids  = (const int*)  d_in[2];
    const float* word_emb  = (const float*)d_in[3];
    const float* swap_emb  = (const float*)d_in[4];
    const float* pos_emb   = (const float*)d_in[5];
    const float* type_emb  = (const float*)d_in[6];
    const float* emb_ln_s  = (const float*)d_in[7];
    const float* emb_ln_b  = (const float*)d_in[8];
    const float* wq = (const float*)d_in[9];   const float* bq = (const float*)d_in[10];
    const float* wk = (const float*)d_in[11];  const float* bk = (const float*)d_in[12];
    const float* wv = (const float*)d_in[13];  const float* bv = (const float*)d_in[14];
    const float* wo = (const float*)d_in[15];  const float* bo = (const float*)d_in[16];
    const float* ln1_s = (const float*)d_in[17]; const float* ln1_b = (const float*)d_in[18];
    const float* w1 = (const float*)d_in[19];  const float* b1 = (const float*)d_in[20];
    const float* w2 = (const float*)d_in[21];  const float* b2 = (const float*)d_in[22];
    const float* ln2_s = (const float*)d_in[23]; const float* ln2_b = (const float*)d_in[24];

    const size_t NT = 2048;        // B*S
    float* ws  = (float*)d_ws;
    float* x   = ws;
    float* x1  = x   + NT * 768;
    float* qb  = x1  + NT * 768;
    float* kbuf= qb  + NT * 768;
    float* vbuf= kbuf+ NT * 768;
    float* cb  = vbuf+ NT * 768;
    float* yb  = cb  + NT * 768;
    float* ffb = yb  + NT * 768;
    ushort_t* wq16 = (ushort_t*)(ffb + NT * 3072);
    const size_t WHH = (size_t)6 * 768 * 768;
    const size_t WHF = (size_t)6 * 768 * 3072;
    ushort_t* wk16 = wq16 + WHH;
    ushort_t* wv16 = wk16 + WHH;
    ushort_t* wo16 = wv16 + WHH;
    ushort_t* w116 = wo16 + WHH;
    ushort_t* w216 = w116 + WHF;

    // 1) convert all GEMM weights to bf16 bits
    cvt_bf16_kernel<<<2048, 256, 0, stream>>>(wq, wq16, (int)WHH);
    cvt_bf16_kernel<<<2048, 256, 0, stream>>>(wk, wk16, (int)WHH);
    cvt_bf16_kernel<<<2048, 256, 0, stream>>>(wv, wv16, (int)WHH);
    cvt_bf16_kernel<<<2048, 256, 0, stream>>>(wo, wo16, (int)WHH);
    cvt_bf16_kernel<<<2048, 256, 0, stream>>>(w1, w116, (int)WHF);
    cvt_bf16_kernel<<<2048, 256, 0, stream>>>(w2, w216, (int)WHF);

    // 2) embeddings + LN
    embed_ln_kernel<<<2048, 256, 0, stream>>>(input_ids, type_ids, word_emb, swap_emb,
                                              pos_emb, type_emb, emb_ln_s, emb_ln_b, x);

    // 3) encoder layers
    const dim3 gH (768 / 128, 2048 / 128);   // N=768 GEMMs
    const dim3 gFF(3072 / 128, 2048 / 128);  // N=3072 GEMM
    for (int l = 0; l < 6; ++l) {
        const ushort_t* Wq = wq16 + (size_t)l * 768 * 768;
        const ushort_t* Wk = wk16 + (size_t)l * 768 * 768;
        const ushort_t* Wv = wv16 + (size_t)l * 768 * 768;
        const ushort_t* Wo = wo16 + (size_t)l * 768 * 768;
        const ushort_t* W1 = w116 + (size_t)l * 768 * 3072;
        const ushort_t* W2 = w216 + (size_t)l * 3072 * 768;

        gemm_bf16_kernel<1><<<gH, 256, 0, stream>>>(x, Wq, bq + (size_t)l * 768, qb,
                                                    2048, 768, 768);
        gemm_bf16_kernel<1><<<gH, 256, 0, stream>>>(x, Wk, bk + (size_t)l * 768, kbuf,
                                                    2048, 768, 768);
        gemm_bf16_kernel<1><<<gH, 256, 0, stream>>>(x, Wv, bv + (size_t)l * 768, vbuf,
                                                    2048, 768, 768);
        attn_kernel<<<4 * 12 * 32, 32, 0, stream>>>(qb, kbuf, vbuf, attn_mask, cb);
        gemm_bf16_kernel<1><<<gH, 256, 0, stream>>>(cb, Wo, bo + (size_t)l * 768, yb,
                                                    2048, 768, 768);
        ln_res_kernel<<<2048, 256, 0, stream>>>(yb, x, ln1_s + (size_t)l * 768,
                                                ln1_b + (size_t)l * 768, x1);
        gemm_bf16_kernel<2><<<gFF, 256, 0, stream>>>(x1, W1, b1 + (size_t)l * 3072, ffb,
                                                     2048, 3072, 768);
        gemm_bf16_kernel<1><<<gH, 256, 0, stream>>>(ffb, W2, b2 + (size_t)l * 768, yb,
                                                    2048, 768, 3072);
        float* out_ptr = (l == 5) ? (float*)d_out : x;
        ln_res_kernel<<<2048, 256, 0, stream>>>(yb, x1, ln2_s + (size_t)l * 768,
                                                ln2_b + (size_t)l * 768, out_ptr);
    }
}
